// SuperPoint_49813030699148
// MI455X (gfx1250) — matrix operand order repaired and checked by reading, not compiled, not run on hardware
//
#include <hip/hip_runtime.h>
#include <hip/hip_bf16.h>

typedef __attribute__((ext_vector_type(16))) _Float16 v16h;
typedef __attribute__((ext_vector_type(8)))  _Float16 v8h;
typedef __attribute__((ext_vector_type(8)))  float    v8f;

#define IMG_B 4
#define IMG_H 480
#define IMG_W 640
#define TOPK  1024
#define BORDER_PX 4
#define SCORE_THR 0.005f
#define EPS_NRM   1e-12f

// ---------------------------------------------------------------- utilities

__global__ void k_zero32(unsigned* __restrict__ p, size_t n) {
  for (size_t i = (size_t)blockIdx.x * blockDim.x + threadIdx.x; i < n;
       i += (size_t)gridDim.x * blockDim.x) p[i] = 0u;
}

// image f32 [b][1][H][W] -> NHWC f16 [b][H][W][CP] with channels 1..CP-1 zeroed
__global__ void k_img_nhwc(const float* __restrict__ img, _Float16* __restrict__ out,
                           size_t npix, int CP) {
  size_t n = npix * (size_t)CP;
  for (size_t i = (size_t)blockIdx.x * blockDim.x + threadIdx.x; i < n;
       i += (size_t)gridDim.x * blockDim.x) {
    int c = (int)(i % (size_t)CP);
    size_t p = i / (size_t)CP;
    out[i] = (c == 0) ? (_Float16)img[p] : (_Float16)0.f;
  }
}

// Pack f32 weights [COUT][CIN][KS][KS] -> f16 [KS*KS][COUTP][CINP], zero pad.
__global__ void k_pack_w(const float* __restrict__ src, _Float16* __restrict__ dst,
                         int COUT, int CIN, int KS, int COUTP, int CINP) {
  size_t n = (size_t)KS * KS * COUTP * CINP;
  for (size_t e = (size_t)blockIdx.x * blockDim.x + threadIdx.x; e < n;
       e += (size_t)gridDim.x * blockDim.x) {
    int ci  = (int)(e % (size_t)CINP);
    size_t r = e / (size_t)CINP;
    int co  = (int)(r % (size_t)COUTP);
    int tap = (int)(r / (size_t)COUTP);
    float v = 0.f;
    if (co < COUT && ci < CIN) v = src[((size_t)co * CIN + ci) * (KS * KS) + tap];
    dst[e] = (_Float16)v;
  }
}

// ---------------------------------------------------------------- WMMA conv
// Implicit GEMM, same-padding KSxKS conv + bias (+ReLU), NHWC activations
// padded to CINP/COUTP channels (padding lanes are kept zero end-to-end).
// Block: 256 threads = 8 waves; tile = 32 cout x 64 px x 4 rows.
// KS==3: input halo tile (ROWS+2 rows x 66 cols x CINP ch, channel-innermost)
//        staged with GLOBAL_LOAD_ASYNC_TO_LDS_B128 (16B granules, contiguous
//        on both sides thanks to NHWC); OOB granules zero-filled via ds store.
// KS==1: B fragment read directly from global (contiguous 32B per lane).
// A frag (16x32 f16): lane L (m=L&15) holds K=16 consecutive channels
//        from ci0+16*(L>>4): two contiguous v8h loads from packed weights.
// B frag (32x16 f16): lane L (n=L&15) holds K=16 consecutive input channels
//        from ci0+16*(L>>4): one contiguous v16h load.
template <int CINP, int COUTP, int KS, bool RELU, bool OUTF32>
__global__ __launch_bounds__(256) void conv_wmma(
    const _Float16* __restrict__ in,    // [nb][H][W][CINP]
    const _Float16* __restrict__ wpk,   // [KS*KS][COUTP][CINP]
    const float*    __restrict__ bias,  // [COUT]
    void* __restrict__ outv,            // [nb][H][W][COUTP] (f16 or f32)
    int COUT, int H, int W) {
  constexpr int ROWS = 4;
  constexpr int TC   = 64 + KS - 1;
  constexpr int TR   = ROWS + KS - 1;
  constexpr int PAD  = KS / 2;
  constexpr int coTiles = COUTP / 32;
  extern __shared__ _Float16 tile[];    // [TR*TC][CINP] (KS==3 only)

  const int b    = blockIdx.z / coTiles;
  const int coB  = (blockIdx.z % coTiles) * 32;
  const int y0   = blockIdx.y * ROWS;
  const int x0   = blockIdx.x * 64;
  const int tid  = threadIdx.x;
  const int lane = tid & 31;
  const int wave = tid >> 5;
  const int hi   = lane >> 4;
  const int lm   = lane & 15;
  const int co0  = coB + 16 * (wave >> 2);
  const int colb = 16 * (wave & 3);

  if (KS == 3) {
    // Async-stage halo tile: 16B (8 f16 channels) per granule.
    constexpr int CH = CINP / 8;
    for (int e = tid; e < TR * TC * CH; e += 256) {
      int c8 = (e % CH) * 8;
      int rc = e / CH;
      int col = rc % TC;
      int row = rc / TC;
      int gy = y0 + row - PAD;
      int gx = x0 + col - PAD;
      unsigned loff = (unsigned)(size_t)&tile[(size_t)rc * CINP + c8];
      if (gy >= 0 && gy < H && gx >= 0 && gx < W) {
        unsigned long long src =
            (unsigned long long)(in + (((size_t)b * H + gy) * W + gx) * CINP + c8);
        asm volatile("global_load_async_to_lds_b128 %0, %1, off"
                     :: "v"(loff), "v"(src) : "memory");
      } else {
        *(uint4*)&tile[(size_t)rc * CINP + c8] = make_uint4(0u, 0u, 0u, 0u);
      }
    }
    asm volatile("s_wait_asynccnt 0" ::: "memory");
    __syncthreads();
  }

  v8f acc[ROWS] = {};
  for (int tap = 0; tap < KS * KS; ++tap) {
    const int trow = tap / KS, kc = tap % KS;
    const _Float16* wrow = wpk + ((size_t)tap * COUTP + (co0 + lm)) * CINP;
    if (tap + 1 < KS * KS)
      __builtin_prefetch(wpk + ((size_t)(tap + 1) * COUTP + (co0 + lm)) * CINP, 0, 0);
    for (int ci0 = 0; ci0 < CINP; ci0 += 32) {
      const v8h a0 = *(const v8h*)(wrow + ci0 + 16 * hi);
      const v8h a1 = *(const v8h*)(wrow + ci0 + 16 * hi + 8);
      v16h af;
#pragma unroll
      for (int t = 0; t < 8; ++t) { af[t] = a0[t]; af[t + 8] = a1[t]; }
#pragma unroll
      for (int r = 0; r < ROWS; ++r) {
        v16h bf;
        if (KS == 3) {
          bf = *(const v16h*)&tile[(size_t)((trow + r) * TC + colb + lm + kc) * CINP +
                                   ci0 + 16 * hi];
        } else {
          int px = x0 + colb + lm;
          if (px > W - 1) px = W - 1;  // clamp: result discarded at store
          bf = *(const v16h*)(in + (((size_t)b * H + (y0 + r)) * W + px) * CINP +
                              ci0 + 16 * hi);
        }
        acc[r] = __builtin_amdgcn_wmma_f32_16x16x32_f16(false, af, false, bf,
                                                        (short)0, acc[r], false, false);
      }
    }
  }

  // D layout: VGPR v -> M = v + 8*(lane>=16), N = lane&15 -> co = co0+8*hi+v.
  const int px = x0 + colb + lm;
  const int cobase = co0 + 8 * hi;
#pragma unroll
  for (int r = 0; r < ROWS; ++r) {
    if (px < W) {
      size_t obase = (((size_t)b * H + (y0 + r)) * W + px) * COUTP + cobase;
      if (OUTF32) {
        float o8[8];
#pragma unroll
        for (int v = 0; v < 8; ++v) {
          float rr = acc[r][v] + bias[cobase + v < COUT ? cobase + v : COUT - 1];
          if (RELU) rr = fmaxf(rr, 0.f);
          o8[v] = rr;
        }
        float* op = (float*)outv + obase;
#pragma unroll
        for (int v = 0; v < 8; ++v)
          if (cobase + v < COUT) op[v] = o8[v];
      } else if (cobase + 8 <= COUT) {
        v8h oh;
#pragma unroll
        for (int v = 0; v < 8; ++v) {
          float rr = acc[r][v] + bias[cobase + v];
          if (RELU) rr = fmaxf(rr, 0.f);
          oh[v] = (_Float16)rr;
        }
        *(v8h*)((_Float16*)outv + obase) = oh;
      } else {
#pragma unroll
        for (int v = 0; v < 8; ++v) {
          if (cobase + v < COUT) {
            float rr = acc[r][v] + bias[cobase + v];
            if (RELU) rr = fmaxf(rr, 0.f);
            ((_Float16*)outv)[obase + v] = (_Float16)rr;
          }
        }
      }
    }
  }
}

// ---------------------------------------------------------------- pool / head

// NHWC 2x2 maxpool
__global__ void k_maxpool2(const _Float16* __restrict__ in, _Float16* __restrict__ out,
                           int NB, int C, int Hi, int Wi) {
  int Ho = Hi / 2, Wo = Wi / 2;
  size_t n = (size_t)NB * Ho * Wo * C;
  for (size_t i = (size_t)blockIdx.x * blockDim.x + threadIdx.x; i < n;
       i += (size_t)gridDim.x * blockDim.x) {
    int c = (int)(i % (size_t)C);
    size_t r = i / (size_t)C;
    int xo = (int)(r % (size_t)Wo);
    size_t r2 = r / (size_t)Wo;
    int yo = (int)(r2 % (size_t)Ho);
    int b = (int)(r2 / (size_t)Ho);
    const _Float16* p = in + (((size_t)b * Hi + 2 * yo) * Wi + 2 * xo) * C + c;
    size_t rs = (size_t)Wi * C;
    float a = (float)p[0], bq = (float)p[C], cq = (float)p[rs], d = (float)p[rs + C];
    out[i] = (_Float16)fmaxf(fmaxf(a, bq), fmaxf(cq, d));
  }
}

// semi NHWC [b][HC][WC][96] f16 -> softmax over 65, drop last, depth-to-space 8x8
__global__ void k_softmax_scores(const _Float16* __restrict__ semi, float* __restrict__ sc,
                                 int NB, int HC, int WC) {
  size_t n = (size_t)NB * HC * WC;
  int Wf = WC * 8;
  for (size_t cell = (size_t)blockIdx.x * blockDim.x + threadIdx.x; cell < n;
       cell += (size_t)gridDim.x * blockDim.x) {
    int wc = (int)(cell % (size_t)WC);
    size_t r = cell / (size_t)WC;
    int hc = (int)(r % (size_t)HC);
    int b  = (int)(r / (size_t)HC);
    const _Float16* lp = semi + cell * 96;
    float l[65], m = -1e30f;
    for (int c = 0; c < 65; ++c) { l[c] = (float)lp[c]; m = fmaxf(m, l[c]); }
    float s = 0.f;
    for (int c = 0; c < 65; ++c) { l[c] = __expf(l[c] - m); s += l[c]; }
    float inv = 1.f / s;
    for (int c = 0; c < 64; ++c)
      sc[((size_t)b * HC * 8 + hc * 8 + (c >> 3)) * Wf + wc * 8 + (c & 7)] = l[c] * inv;
  }
}

// ---------------------------------------------------------------- NMS (9x9 separable)

__global__ void k_rowmax9(const float* __restrict__ in, float* __restrict__ out,
                          int NP, int H, int W) {
  size_t n = (size_t)NP * H * W;
  for (size_t i = (size_t)blockIdx.x * blockDim.x + threadIdx.x; i < n;
       i += (size_t)gridDim.x * blockDim.x) {
    int x = (int)(i % (size_t)W);
    const float* row = in + (i - x);
    int lo = x - 4 > 0 ? x - 4 : 0;
    int hh = x + 4 < W - 1 ? x + 4 : W - 1;
    float m = -1e30f;
    for (int xx = lo; xx <= hh; ++xx) m = fmaxf(m, row[xx]);
    out[i] = m;
  }
}

__global__ void k_colmax9(const float* __restrict__ in, float* __restrict__ out,
                          int NP, int H, int W) {
  size_t n = (size_t)NP * H * W;
  for (size_t i = (size_t)blockIdx.x * blockDim.x + threadIdx.x; i < n;
       i += (size_t)gridDim.x * blockDim.x) {
    int x = (int)(i % (size_t)W);
    size_t r = i / (size_t)W;
    int y = (int)(r % (size_t)H);
    size_t p = r / (size_t)H;
    int lo = y - 4 > 0 ? y - 4 : 0;
    int hh = y + 4 < H - 1 ? y + 4 : H - 1;
    float m = -1e30f;
    const float* pl = in + (size_t)p * H * W + x;
    for (int yy = lo; yy <= hh; ++yy) m = fmaxf(m, pl[(size_t)yy * W]);
    out[i] = m;
  }
}

__global__ void k_eqmask(const float* __restrict__ s, const float* __restrict__ mp,
                         float* __restrict__ mask, size_t n) {
  for (size_t i = (size_t)blockIdx.x * blockDim.x + threadIdx.x; i < n;
       i += (size_t)gridDim.x * blockDim.x) mask[i] = (s[i] == mp[i]) ? 1.f : 0.f;
}

__global__ void k_supp(const float* __restrict__ s, const float* __restrict__ mpm,
                       float* __restrict__ out, size_t n) {
  for (size_t i = (size_t)blockIdx.x * blockDim.x + threadIdx.x; i < n;
       i += (size_t)gridDim.x * blockDim.x) out[i] = (mpm[i] > 0.f) ? 0.f : s[i];
}

__global__ void k_update(float* __restrict__ mask, const float* __restrict__ ss,
                         const float* __restrict__ mpm, const float* __restrict__ mpss,
                         size_t n) {
  for (size_t i = (size_t)blockIdx.x * blockDim.x + threadIdx.x; i < n;
       i += (size_t)gridDim.x * blockDim.x) {
    bool supp = mpm[i] > 0.f;
    bool nm = (ss[i] == mpss[i]);
    if (nm && !supp) mask[i] = 1.f;
  }
}

__global__ void k_nmsout(const float* __restrict__ s, const float* __restrict__ mask,
                         float* __restrict__ out, size_t n) {
  for (size_t i = (size_t)blockIdx.x * blockDim.x + threadIdx.x; i < n;
       i += (size_t)gridDim.x * blockDim.x) out[i] = (mask[i] != 0.f) ? s[i] : 0.f;
}

// ---------------------------------------------------------------- top-k

__global__ void k_hist(const float* __restrict__ nms, unsigned* __restrict__ hist,
                       int NB, int H, int W, int border) {
  size_t n = (size_t)NB * H * W;
  for (size_t i = (size_t)blockIdx.x * blockDim.x + threadIdx.x; i < n;
       i += (size_t)gridDim.x * blockDim.x) {
    int x = (int)(i % (size_t)W);
    size_t r = i / (size_t)W;
    int y = (int)(r % (size_t)H);
    int b = (int)(r / (size_t)H);
    float v = nms[i];
    if (!(x >= border && x < W - border && y >= border && y < H - border)) v = 0.f;
    unsigned bin = __float_as_uint(v) >> 21;   // non-negative floats: monotonic
    atomicAdd(&hist[(size_t)b * 2048 + bin], 1u);
  }
}

__global__ void k_cutoff(const unsigned* __restrict__ hist, int* __restrict__ cut,
                         int NB, int K) {
  int b = blockIdx.x * blockDim.x + threadIdx.x;
  if (b < NB) {
    unsigned acc = 0; int c = 0;
    for (int bin = 2047; bin >= 0; --bin) {
      acc += hist[(size_t)b * 2048 + bin];
      if (acc >= (unsigned)K) { c = bin; break; }
    }
    cut[b] = c;
  }
}

__global__ void k_compact(const float* __restrict__ nms, const int* __restrict__ cut,
                          float* __restrict__ cval, int* __restrict__ cidx,
                          int* __restrict__ cnt, int NB, int H, int W, int border) {
  size_t n = (size_t)NB * H * W;
  for (size_t i = (size_t)blockIdx.x * blockDim.x + threadIdx.x; i < n;
       i += (size_t)gridDim.x * blockDim.x) {
    int x = (int)(i % (size_t)W);
    size_t r = i / (size_t)W;
    int y = (int)(r % (size_t)H);
    int b = (int)(r / (size_t)H);
    float v = nms[i];
    if (!(x >= border && x < W - border && y >= border && y < H - border)) v = 0.f;
    int bin = (int)(__float_as_uint(v) >> 21);
    int c = cut[b];
    int pix = y * W + x;
    if (bin > c) {                       // strictly above cutoff: < K entries
      int s_ = atomicAdd(&cnt[b * 2 + 0], 1);
      if (s_ < 1024) { cval[(size_t)b * 4096 + s_] = v; cidx[(size_t)b * 4096 + s_] = pix; }
    } else if (bin == c) {               // cutoff bin: capped tie region
      int s_ = atomicAdd(&cnt[b * 2 + 1], 1);
      if (s_ < 3072) { cval[(size_t)b * 4096 + 1024 + s_] = v; cidx[(size_t)b * 4096 + 1024 + s_] = pix; }
    }
  }
}

__global__ __launch_bounds__(1024) void k_sort_select(
    const float* __restrict__ cval, const int* __restrict__ cidx,
    const int* __restrict__ cnt, float* __restrict__ kp, float* __restrict__ ks,
    int K, int W) {
  __shared__ float sv[4096];
  __shared__ int   si[4096];
  int b = blockIdx.x;
  int nHi = cnt[b * 2 + 0]; if (nHi > 1024) nHi = 1024;
  int nEq = cnt[b * 2 + 1]; if (nEq > 3072) nEq = 3072;
  for (int i = threadIdx.x; i < 4096; i += 1024) {
    bool valid = (i < 1024) ? (i < nHi) : ((i - 1024) < nEq);
    sv[i] = valid ? cval[(size_t)b * 4096 + i] : -1e30f;
    si[i] = valid ? cidx[(size_t)b * 4096 + i] : 0x7fffffff;
  }
  __syncthreads();
  // Bitonic sort, descending by value, ascending by index on ties.
  for (int kk = 2; kk <= 4096; kk <<= 1) {
    for (int j = kk >> 1; j > 0; j >>= 1) {
      for (int i = threadIdx.x; i < 4096; i += 1024) {
        int ixj = i ^ j;
        if (ixj > i) {
          float va = sv[i], vb = sv[ixj];
          int ia = si[i], ib = si[ixj];
          bool beforeAB = (va > vb) || (va == vb && ia < ib);
          bool descBlock = ((i & kk) == 0);
          bool doswap = descBlock ? !beforeAB : beforeAB;
          if (doswap) { sv[i] = vb; sv[ixj] = va; si[i] = ib; si[ixj] = ia; }
        }
      }
      __syncthreads();
    }
  }
  for (int i = threadIdx.x; i < K; i += 1024) {
    float v = sv[i];
    int id = si[i];
    int ky = id / W, kx = id - ky * W;
    kp[((size_t)b * K + i) * 2 + 0] = (float)kx;
    kp[((size_t)b * K + i) * 2 + 1] = (float)ky;
    ks[(size_t)b * K + i] = (v > SCORE_THR) ? v : 0.f;
  }
}

// ---------------------------------------------------------------- descriptors

// dmap NHWC f32 [P][256]; per-pixel L2 normalize in place
__global__ __launch_bounds__(256) void k_norm_desc_map(float* __restrict__ dm, int C, int P) {
  __shared__ float red[256];
  int p = blockIdx.x, c = threadIdx.x;
  float v = dm[(size_t)p * C + c];
  red[c] = v * v;
  __syncthreads();
  for (int s2 = 128; s2 > 0; s2 >>= 1) {
    if (c < s2) red[c] += red[c + s2];
    __syncthreads();
  }
  float inv = 1.f / fmaxf(sqrtf(red[0]), EPS_NRM);
  dm[(size_t)p * C + c] = v * inv;
}

__global__ __launch_bounds__(256) void k_sample_desc(
    const float* __restrict__ dm, const float* __restrict__ kp0,
    float* __restrict__ outd, int C, int h, int w, int K) {
  __shared__ float red[256];
  int k = blockIdx.x, c = threadIdx.x;
  float kx = kp0[k * 2 + 0], ky = kp0[k * 2 + 1];
  float gx = ((kx - 3.5f) / ((float)(w * 8) - 4.5f)) * 2.f - 1.f;
  float gy = ((ky - 3.5f) / ((float)(h * 8) - 4.5f)) * 2.f - 1.f;
  float ix = (gx + 1.f) * 0.5f * (float)(w - 1);
  float iy = (gy + 1.f) * 0.5f * (float)(h - 1);
  float x0 = floorf(ix), y0 = floorf(iy);
  float wx1 = ix - x0, wx0 = 1.f - wx1, wy1 = iy - y0, wy0 = 1.f - wy1;
  auto g = [&](float yi, float xi) -> float {
    bool valid = (xi >= 0.f) && (xi <= (float)(w - 1)) && (yi >= 0.f) && (yi <= (float)(h - 1));
    int xc = (int)fminf(fmaxf(xi, 0.f), (float)(w - 1));
    int yc = (int)fminf(fmaxf(yi, 0.f), (float)(h - 1));
    return valid ? dm[((size_t)yc * w + xc) * C + c] : 0.f;
  };
  float v = g(y0, x0) * wy0 * wx0 + g(y0, x0 + 1.f) * wy0 * wx1 +
            g(y0 + 1.f, x0) * wy1 * wx0 + g(y0 + 1.f, x0 + 1.f) * wy1 * wx1;
  red[c] = v * v;
  __syncthreads();
  for (int s2 = 128; s2 > 0; s2 >>= 1) {
    if (c < s2) red[c] += red[c + s2];
    __syncthreads();
  }
  outd[(size_t)c * K + k] = v / fmaxf(sqrtf(red[0]), EPS_NRM);
}

// ---------------------------------------------------------------- launch

static inline int nblk(size_t n) {
  size_t b = (n + 255) / 256;
  return (int)(b > 4096 ? 4096 : (b == 0 ? 1 : b));
}

extern "C" void kernel_launch(void* const* d_in, const int* in_sizes, int n_in,
                              void* d_out, int out_size, void* d_ws, size_t ws_size,
                              hipStream_t stream) {
  (void)in_sizes; (void)n_in; (void)out_size; (void)ws_size;
  const float* image = (const float*)d_in[0];
  const float* w1a = (const float*)d_in[1];  const float* b1a = (const float*)d_in[2];
  const float* w1b = (const float*)d_in[3];  const float* b1b = (const float*)d_in[4];
  const float* w2a = (const float*)d_in[5];  const float* b2a = (const float*)d_in[6];
  const float* w2b = (const float*)d_in[7];  const float* b2b = (const float*)d_in[8];
  const float* w3a = (const float*)d_in[9];  const float* b3a = (const float*)d_in[10];
  const float* w3b = (const float*)d_in[11]; const float* b3b = (const float*)d_in[12];
  const float* w4a = (const float*)d_in[13]; const float* b4a = (const float*)d_in[14];
  const float* w4b = (const float*)d_in[15]; const float* b4b = (const float*)d_in[16];
  const float* wPa = (const float*)d_in[17]; const float* bPa = (const float*)d_in[18];
  const float* wPb = (const float*)d_in[19]; const float* bPb = (const float*)d_in[20];
  const float* wDa = (const float*)d_in[21]; const float* bDa = (const float*)d_in[22];
  const float* wDb = (const float*)d_in[23]; const float* bDb = (const float*)d_in[24];

  float* out      = (float*)d_out;
  float* out_nms  = out;                                             // 4*480*640
  float* out_kp   = out + (size_t)IMG_B * IMG_H * IMG_W;             // 4*1024*2
  float* out_ks   = out_kp + (size_t)IMG_B * TOPK * 2;               // 4*1024
  float* out_desc = out_ks + (size_t)IMG_B * TOPK;                   // 256*1024

  // ---- workspace carve-out (~360 MB assumed available)
  char* base = (char*)d_ws;
  size_t off = 0;
  auto alloc = [&](size_t bytes) -> void* {
    void* p = base + off;
    off = (off + bytes + 255) & ~(size_t)255;
    return p;
  };
  const size_t actMax = (size_t)IMG_B * IMG_H * IMG_W * 64;          // largest f16 layer
  _Float16* bufA = (_Float16*)alloc(actMax * 2);
  _Float16* bufB = (_Float16*)alloc(actMax * 2);
  _Float16* semi = (_Float16*)alloc((size_t)IMG_B * 60 * 80 * 96 * 2);
  const size_t nsp = (size_t)IMG_B * IMG_H * IMG_W;
  float* scores = (float*)alloc(nsp * 4);
  float* maskb  = (float*)alloc(nsp * 4);
  float* t0 = (float*)alloc(nsp * 4);
  float* t1 = (float*)alloc(nsp * 4);
  float* t2 = (float*)alloc(nsp * 4);
  float* t3 = (float*)alloc(nsp * 4);
  float* dmap = (float*)alloc((size_t)60 * 80 * 256 * 4);
  unsigned* hist = (unsigned*)alloc((size_t)IMG_B * 2048 * 4);
  int* cnt = (int*)alloc((size_t)IMG_B * 2 * 4);
  int* cut = (int*)alloc((size_t)IMG_B * 4);
  float* cval = (float*)alloc((size_t)IMG_B * 4096 * 4);
  int*   cidx = (int*)alloc((size_t)IMG_B * 4096 * 4);
  _Float16* pw1a = (_Float16*)alloc((size_t)9 * 64 * 32 * 2);
  _Float16* pw1b = (_Float16*)alloc((size_t)9 * 64 * 64 * 2);
  _Float16* pw2a = (_Float16*)alloc((size_t)9 * 64 * 64 * 2);
  _Float16* pw2b = (_Float16*)alloc((size_t)9 * 64 * 64 * 2);
  _Float16* pw3a = (_Float16*)alloc((size_t)9 * 128 * 64 * 2);
  _Float16* pw3b = (_Float16*)alloc((size_t)9 * 128 * 128 * 2);
  _Float16* pw4a = (_Float16*)alloc((size_t)9 * 128 * 128 * 2);
  _Float16* pw4b = (_Float16*)alloc((size_t)9 * 128 * 128 * 2);
  _Float16* pwPa = (_Float16*)alloc((size_t)9 * 256 * 128 * 2);
  _Float16* pwPb = (_Float16*)alloc((size_t)96 * 256 * 2);
  _Float16* pwDa = (_Float16*)alloc((size_t)9 * 256 * 128 * 2);
  _Float16* pwDb = (_Float16*)alloc((size_t)256 * 256 * 2);

  // ---- weight packing + image conversion (NHWC, 32-ch zero-padded)
  k_pack_w<<<nblk(9 * 64 * 32), 256, 0, stream>>>(w1a, pw1a, 64, 1, 3, 64, 32);
  k_pack_w<<<nblk(9 * 64 * 64), 256, 0, stream>>>(w1b, pw1b, 64, 64, 3, 64, 64);
  k_pack_w<<<nblk(9 * 64 * 64), 256, 0, stream>>>(w2a, pw2a, 64, 64, 3, 64, 64);
  k_pack_w<<<nblk(9 * 64 * 64), 256, 0, stream>>>(w2b, pw2b, 64, 64, 3, 64, 64);
  k_pack_w<<<nblk(9 * 128 * 64), 256, 0, stream>>>(w3a, pw3a, 128, 64, 3, 128, 64);
  k_pack_w<<<nblk(9 * 128 * 128), 256, 0, stream>>>(w3b, pw3b, 128, 128, 3, 128, 128);
  k_pack_w<<<nblk(9 * 128 * 128), 256, 0, stream>>>(w4a, pw4a, 128, 128, 3, 128, 128);
  k_pack_w<<<nblk(9 * 128 * 128), 256, 0, stream>>>(w4b, pw4b, 128, 128, 3, 128, 128);
  k_pack_w<<<nblk(9 * 256 * 128), 256, 0, stream>>>(wPa, pwPa, 256, 128, 3, 256, 128);
  k_pack_w<<<nblk(96 * 256), 256, 0, stream>>>(wPb, pwPb, 65, 256, 1, 96, 256);
  k_pack_w<<<nblk(9 * 256 * 128), 256, 0, stream>>>(wDa, pwDa, 256, 128, 3, 256, 128);
  k_pack_w<<<nblk(256 * 256), 256, 0, stream>>>(wDb, pwDb, 256, 256, 1, 256, 256);
  k_img_nhwc<<<nblk(nsp * 32), 256, 0, stream>>>(image, bufA, nsp, 32);

  // ---- backbone (ping-pong bufA/bufB), dynamic LDS = (ROWS+2)*66*CINP f16
  dim3 blk(256);
  conv_wmma<32, 64, 3, true, false><<<dim3(10, 120, IMG_B * 2), blk, 6 * 66 * 32 * 2, stream>>>(
      bufA, pw1a, b1a, bufB, 64, 480, 640);
  conv_wmma<64, 64, 3, true, false><<<dim3(10, 120, IMG_B * 2), blk, 6 * 66 * 64 * 2, stream>>>(
      bufB, pw1b, b1b, bufA, 64, 480, 640);
  k_maxpool2<<<nblk((size_t)IMG_B * 64 * 240 * 320), 256, 0, stream>>>(bufA, bufB, IMG_B, 64, 480, 640);
  conv_wmma<64, 64, 3, true, false><<<dim3(5, 60, IMG_B * 2), blk, 6 * 66 * 64 * 2, stream>>>(
      bufB, pw2a, b2a, bufA, 64, 240, 320);
  conv_wmma<64, 64, 3, true, false><<<dim3(5, 60, IMG_B * 2), blk, 6 * 66 * 64 * 2, stream>>>(
      bufA, pw2b, b2b, bufB, 64, 240, 320);
  k_maxpool2<<<nblk((size_t)IMG_B * 64 * 120 * 160), 256, 0, stream>>>(bufB, bufA, IMG_B, 64, 240, 320);
  conv_wmma<64, 128, 3, true, false><<<dim3(3, 30, IMG_B * 4), blk, 6 * 66 * 64 * 2, stream>>>(
      bufA, pw3a, b3a, bufB, 128, 120, 160);
  conv_wmma<128, 128, 3, true, false><<<dim3(3, 30, IMG_B * 4), blk, 6 * 66 * 128 * 2, stream>>>(
      bufB, pw3b, b3b, bufA, 128, 120, 160);
  k_maxpool2<<<nblk((size_t)IMG_B * 128 * 60 * 80), 256, 0, stream>>>(bufA, bufB, IMG_B, 128, 120, 160);
  conv_wmma<128, 128, 3, true, false><<<dim3(2, 15, IMG_B * 4), blk, 6 * 66 * 128 * 2, stream>>>(
      bufB, pw4a, b4a, bufA, 128, 60, 80);
  conv_wmma<128, 128, 3, true, false><<<dim3(2, 15, IMG_B * 4), blk, 6 * 66 * 128 * 2, stream>>>(
      bufA, pw4b, b4b, bufB, 128, 60, 80);                       // x = bufB

  // ---- keypoint head
  conv_wmma<128, 256, 3, true, false><<<dim3(2, 15, IMG_B * 8), blk, 6 * 66 * 128 * 2, stream>>>(
      bufB, pwPa, bPa, bufA, 256, 60, 80);
  conv_wmma<256, 96, 1, false, false><<<dim3(2, 15, IMG_B * 3), blk, 0, stream>>>(
      bufA, pwPb, bPb, semi, 65, 60, 80);
  k_softmax_scores<<<nblk((size_t)IMG_B * 60 * 80), 256, 0, stream>>>(semi, scores, IMG_B, 60, 80);

  // ---- NMS (2-iteration suppression, 9x9 separable max)
  int nb_sp = nblk(nsp);
  k_rowmax9<<<nb_sp, 256, 0, stream>>>(scores, t0, IMG_B, IMG_H, IMG_W);
  k_colmax9<<<nb_sp, 256, 0, stream>>>(t0, t1, IMG_B, IMG_H, IMG_W);
  k_eqmask<<<nb_sp, 256, 0, stream>>>(scores, t1, maskb, nsp);
  for (int it = 0; it < 2; ++it) {
    k_rowmax9<<<nb_sp, 256, 0, stream>>>(maskb, t0, IMG_B, IMG_H, IMG_W);
    k_colmax9<<<nb_sp, 256, 0, stream>>>(t0, t1, IMG_B, IMG_H, IMG_W);   // t1 = mp(mask)
    k_supp<<<nb_sp, 256, 0, stream>>>(scores, t1, t0, nsp);              // t0 = supp_scores
    k_rowmax9<<<nb_sp, 256, 0, stream>>>(t0, t2, IMG_B, IMG_H, IMG_W);
    k_colmax9<<<nb_sp, 256, 0, stream>>>(t2, t3, IMG_B, IMG_H, IMG_W);   // t3 = mp(supp_scores)
    k_update<<<nb_sp, 256, 0, stream>>>(maskb, t0, t1, t3, nsp);
  }
  k_nmsout<<<nb_sp, 256, 0, stream>>>(scores, maskb, out_nms, nsp);

  // ---- top-K selection
  k_zero32<<<nblk(IMG_B * 2048), 256, 0, stream>>>(hist, (size_t)IMG_B * 2048);
  k_zero32<<<1, 256, 0, stream>>>((unsigned*)cnt, (size_t)IMG_B * 2);
  k_hist<<<nb_sp, 256, 0, stream>>>(out_nms, hist, IMG_B, IMG_H, IMG_W, BORDER_PX);
  k_cutoff<<<1, 32, 0, stream>>>(hist, cut, IMG_B, TOPK);
  k_compact<<<nb_sp, 256, 0, stream>>>(out_nms, cut, cval, cidx, cnt, IMG_B, IMG_H, IMG_W, BORDER_PX);
  k_sort_select<<<IMG_B, 1024, 0, stream>>>(cval, cidx, cnt, out_kp, out_ks, TOPK, IMG_W);

  // ---- descriptor branch (batch 0 only)
  conv_wmma<128, 256, 3, true, false><<<dim3(2, 15, 1 * 8), blk, 6 * 66 * 128 * 2, stream>>>(
      bufB, pwDa, bDa, bufA, 256, 60, 80);
  conv_wmma<256, 256, 1, false, true><<<dim3(2, 15, 1 * 8), blk, 0, stream>>>(
      bufA, pwDb, bDb, dmap, 256, 60, 80);
  k_norm_desc_map<<<60 * 80, 256, 0, stream>>>(dmap, 256, 60 * 80);
  k_sample_desc<<<TOPK, 256, 0, stream>>>(dmap, out_kp, out_desc, 256, 60, 80, TOPK);
}
